// LSTMTaggerCharacter_25297357373499
// MI455X (gfx1250) — matrix operand order repaired and checked by reading, not compiled, not run on hardware
//
#include <hip/hip_runtime.h>

// ---------------- problem dims (from the reference) ----------------
#define DS   2048           // sentence length S
#define DL   12             // chars per word L
#define DSL  (DS*DL)        // 24576
#define DEC  64             // char embed dim Ec
#define DHC  256            // char hidden Hc
#define G4HC (4*DHC)        // 1024
#define DE   512            // word embed dim E
#define DEHC (DE+DHC)       // 768
#define DH   1024           // word hidden H
#define G4H  (4*DH)         // 4096
#define DT   128            // tagset T

typedef __bf16 bf16x8  __attribute__((ext_vector_type(8)));
typedef __bf16 bf16x16 __attribute__((ext_vector_type(16)));
typedef float  f32x8   __attribute__((ext_vector_type(8)));
typedef int    v4i     __attribute__((ext_vector_type(4)));

// ---------------- CDNA5 async copy to LDS (ASYNCcnt-tracked) ----------------
__device__ __forceinline__ void async_copy_b128(const __bf16* g, __bf16* l) {
#if __has_builtin(__builtin_amdgcn_global_load_async_to_lds_b128)
    __builtin_amdgcn_global_load_async_to_lds_b128(
        (__attribute__((address_space(1))) v4i*)(void*)g,
        (__attribute__((address_space(3))) v4i*)(void*)l, 0, 0);
#else
    *(bf16x8*)l = *(const bf16x8*)g;
#endif
}
template <int N>
__device__ __forceinline__ void wait_async() {
#if __has_builtin(__builtin_amdgcn_s_wait_asynccnt)
    __builtin_amdgcn_s_wait_asynccnt(N);
#endif
}

// ---------------- small helpers ----------------
__global__ void k_convert_bf16(const float* __restrict__ in, __bf16* __restrict__ out, long long n) {
    long long i = (long long)blockIdx.x * blockDim.x + threadIdx.x;
    long long stride = (long long)gridDim.x * blockDim.x;
    for (; i < n; i += stride) out[i] = (__bf16)in[i];
}

__global__ void k_bias_sum(const float* __restrict__ a, const float* __restrict__ b,
                           float* __restrict__ out, int n) {
    int i = blockIdx.x * blockDim.x + threadIdx.x;
    if (i < n) out[i] = a[i] + b[i];
}

// gather char embeddings -> bf16 [SL, Ec]
__global__ void k_gather_char(const int* __restrict__ wc, const float* __restrict__ cemb,
                              __bf16* __restrict__ ce) {
    long long i = (long long)blockIdx.x * blockDim.x + threadIdx.x;
    if (i >= (long long)DSL * DEC) return;
    int row = (int)(i / DEC), col = (int)(i % DEC);
    ce[i] = (__bf16)cemb[(long long)wc[row] * DEC + col];
}

// gather word embeddings into embeds[:, 0:512] (bf16 [S, E+Hc])
__global__ void k_gather_word(const int* __restrict__ sent, const float* __restrict__ wemb,
                              __bf16* __restrict__ embeds) {
    long long i = (long long)blockIdx.x * blockDim.x + threadIdx.x;
    if (i >= (long long)DS * DE) return;
    int s = (int)(i / DE), e = (int)(i % DE);
    embeds[(long long)s * DEHC + e] = (__bf16)wemb[(long long)sent[s] * DE + e];
}

// ---------------- WMMA GEMM: C[M,N] = A[M,K](bf16) x B[N,K](bf16)^T + bias[N] ----------------
// B is stored row-major [N,K] (exactly the weight of x @ W.T).
// Block = 128 threads = 4 waves; each wave computes a 16(M) x 64(N) tile.
// The 64x32 B slab is shared by all 4 waves -> double-buffered in LDS via
// async-to-LDS loads. Requires K % 32 == 0, N % 64 == 0, M % 64 == 0.
#define BROW 40   // LDS row pitch in bf16 elems (80B) -> conflict-free half-wave reads
__global__ __launch_bounds__(128)
void k_gemm_bf16(const __bf16* __restrict__ A, const __bf16* __restrict__ B,
                 const float* __restrict__ bias, float* __restrict__ C,
                 int M, int N, int K) {
    __shared__ __align__(16) __bf16 Bs[2][64 * BROW];
    const int tid  = threadIdx.x;
    const int wave = tid >> 5;
    const int lane = tid & 31;
    const int half = lane >> 4;     // which 16-lane half of the wave
    const int lr   = lane & 15;
    const int m0 = (blockIdx.y * 4 + wave) * 16;
    const int n0 = blockIdx.x * 64;

    // staging map: thread t copies 32B of row n0 + t/2, halves selected by t&1
    const int srow = tid >> 1;
    const int scol = (tid & 1) * 16;
    const __bf16* bsrc = B + (size_t)(n0 + srow) * K + scol;
    const int ldst = srow * BROW + scol;
    const int ksteps = K / 32;

    // prologue: stage k-slab 0 into buffer 0
    async_copy_b128(bsrc + 0, &Bs[0][ldst + 0]);
    async_copy_b128(bsrc + 8, &Bs[0][ldst + 8]);

    const int mrow = (m0 + lr < M) ? (m0 + lr) : (M - 1);  // clamp (M%64==0 here anyway)
    const __bf16* arow = A + (size_t)mrow * K;

    f32x8 acc[4] = {};
    for (int ks = 0; ks < ksteps; ++ks) {
        const int k0 = ks * 32;
        // A fragment (16x32 bf16): lanes 0-15 hold K=k0+[0..7],k0+16+[0..7];
        //                          lanes 16-31 hold K=k0+8+[0..7],k0+24+[0..7].
        bf16x8 alo = *(const bf16x8*)(arow + k0 + half * 8);
        bf16x8 ahi = *(const bf16x8*)(arow + k0 + 16 + half * 8);
        if (ks + 1 < ksteps) {           // stage next slab into the other buffer
            const __bf16* s = bsrc + (ks + 1) * 32;
            __bf16* d = &Bs[(ks + 1) & 1][ldst];
            async_copy_b128(s + 0, d + 0);
            async_copy_b128(s + 8, d + 8);
            wait_async<2>();             // slab ks complete (in-order), 2 newest may fly
        } else {
            wait_async<0>();
        }
        __syncthreads();

        bf16x16 a;
#pragma unroll
        for (int i = 0; i < 8; ++i) { a[i] = alo[i]; a[8 + i] = ahi[i]; }
        const __bf16* bbuf = Bs[ks & 1];
#pragma unroll
        for (int j = 0; j < 4; ++j) {
            // B fragment (32x16): col n = j*16+lr, lanes 0-15 K=k0+[0..7],k0+16+[0..7]; 16-31 K=k0+8+[0..7],k0+24+[0..7]
            const __bf16* bf = bbuf + (j * 16 + lr) * BROW + half * 8;
            bf16x8 blo = *(const bf16x8*)(bf);
            bf16x8 bhi = *(const bf16x8*)(bf + 16);
            bf16x16 b;
#pragma unroll
            for (int i = 0; i < 8; ++i) { b[i] = blo[i]; b[8 + i] = bhi[i]; }
            acc[j] = __builtin_amdgcn_wmma_f32_16x16x32_bf16(
                false, a, false, b, (short)0, acc[j], false, false);
        }
        __syncthreads();                 // reads done before next-next stage overwrites
    }

    if (m0 < M) {
        // D layout: VGPR v holds row m0+v (lanes 0-15) / m0+8+v (lanes 16-31)
#pragma unroll
        for (int j = 0; j < 4; ++j) {
            const int col = n0 + j * 16 + lr;
            const float bv = bias ? bias[col] : 0.0f;
            float* cbase = C + (size_t)(m0 + half * 8) * N + col;
#pragma unroll
            for (int v = 0; v < 8; ++v)
                cbase[(size_t)v * N] = acc[j][v] + bv;
        }
    }
}

// ---------------- device math ----------------
__device__ __forceinline__ float sigm(float x) { return 1.0f / (1.0f + __expf(-x)); }

// ---------------- char LSTM scan (persistent single workgroup) ----------------
// 1024 threads; thread tid computes gate row tid of Whh_c (1024x256, bf16) each step.
// Gate order i,f,g,o. Writes h at each word's last char into embeds[:, 512:768].
__global__ __launch_bounds__(1024)
void k_char_lstm(const float* __restrict__ xpre,    // [SL, 1024], biases pre-folded
                 const __bf16* __restrict__ Whh,    // [1024, 256] bf16
                 __bf16* __restrict__ embeds) {     // [S, 768]
    __shared__ __align__(16) float h_lds[DHC];
    __shared__ float gates[G4HC];
    const int tid = threadIdx.x;
    if (tid < DHC) h_lds[tid] = 0.0f;
    float c = 0.0f;
    __syncthreads();

    const bf16x8* wrow = (const bf16x8*)(Whh + (size_t)tid * DHC);   // 32 chunks of 8
    for (int t = 0; t < DSL; ++t) {
        float acc = xpre[(size_t)t * G4HC + tid];
        if (t + 1 < DSL) __builtin_prefetch(xpre + (size_t)(t + 1) * G4HC + tid, 0, 0);
#pragma unroll 4
        for (int k = 0; k < DHC / 8; ++k) {
            bf16x8 w = wrow[k];
            float4 h0 = ((const float4*)h_lds)[2 * k];
            float4 h1 = ((const float4*)h_lds)[2 * k + 1];
            acc = fmaf((float)w[0], h0.x, acc);
            acc = fmaf((float)w[1], h0.y, acc);
            acc = fmaf((float)w[2], h0.z, acc);
            acc = fmaf((float)w[3], h0.w, acc);
            acc = fmaf((float)w[4], h1.x, acc);
            acc = fmaf((float)w[5], h1.y, acc);
            acc = fmaf((float)w[6], h1.z, acc);
            acc = fmaf((float)w[7], h1.w, acc);
        }
        gates[tid] = acc;
        __syncthreads();
        if (tid < DHC) {
            float gi = sigm(gates[tid]);
            float gf = sigm(gates[DHC + tid]);
            float gg = tanhf(gates[2 * DHC + tid]);
            float go = sigm(gates[3 * DHC + tid]);
            c = gf * c + gi * gg;
            float h = go * tanhf(c);
            h_lds[tid] = h;
            if ((t % DL) == DL - 1) {
                int s = t / DL;
                embeds[(size_t)s * DEHC + DE + tid] = (__bf16)h;
            }
        }
        __syncthreads();
    }
}

// ---------------- word LSTM scan (persistent single workgroup) ----------------
// 1024 threads; thread tid owns hidden unit tid, computes gate rows
// (tid, 1024+tid, 2048+tid, 3072+tid) of Whh_t (4096x1024, bf16) each step.
__global__ __launch_bounds__(1024)
void k_word_lstm(const float* __restrict__ xpre,    // [S, 4096], biases pre-folded
                 const __bf16* __restrict__ Whh,    // [4096, 1024] bf16
                 __bf16* __restrict__ hsb) {        // [S, 1024] (bf16 for the out-GEMM)
    __shared__ __align__(16) float h_lds[DH];
    const int tid = threadIdx.x;
    h_lds[tid] = 0.0f;
    float c = 0.0f;
    __syncthreads();

    const bf16x8* wi = (const bf16x8*)(Whh + (size_t)(0 * DH + tid) * DH);
    const bf16x8* wf = (const bf16x8*)(Whh + (size_t)(1 * DH + tid) * DH);
    const bf16x8* wg = (const bf16x8*)(Whh + (size_t)(2 * DH + tid) * DH);
    const bf16x8* wo = (const bf16x8*)(Whh + (size_t)(3 * DH + tid) * DH);

    for (int t = 0; t < DS; ++t) {
        const float* xr = xpre + (size_t)t * G4H;
        float ai = xr[tid], af = xr[DH + tid], ag = xr[2 * DH + tid], ao = xr[3 * DH + tid];
        if (t + 1 < DS) __builtin_prefetch(xpre + (size_t)(t + 1) * G4H + tid, 0, 0);
#pragma unroll 2
        for (int k = 0; k < DH / 8; ++k) {
            float4 h0 = ((const float4*)h_lds)[2 * k];
            float4 h1 = ((const float4*)h_lds)[2 * k + 1];
            bf16x8 a = wi[k], b = wf[k], g = wg[k], o = wo[k];
            ai = fmaf((float)a[0], h0.x, ai); ai = fmaf((float)a[1], h0.y, ai);
            ai = fmaf((float)a[2], h0.z, ai); ai = fmaf((float)a[3], h0.w, ai);
            ai = fmaf((float)a[4], h1.x, ai); ai = fmaf((float)a[5], h1.y, ai);
            ai = fmaf((float)a[6], h1.z, ai); ai = fmaf((float)a[7], h1.w, ai);
            af = fmaf((float)b[0], h0.x, af); af = fmaf((float)b[1], h0.y, af);
            af = fmaf((float)b[2], h0.z, af); af = fmaf((float)b[3], h0.w, af);
            af = fmaf((float)b[4], h1.x, af); af = fmaf((float)b[5], h1.y, af);
            af = fmaf((float)b[6], h1.z, af); af = fmaf((float)b[7], h1.w, af);
            ag = fmaf((float)g[0], h0.x, ag); ag = fmaf((float)g[1], h0.y, ag);
            ag = fmaf((float)g[2], h0.z, ag); ag = fmaf((float)g[3], h0.w, ag);
            ag = fmaf((float)g[4], h1.x, ag); ag = fmaf((float)g[5], h1.y, ag);
            ag = fmaf((float)g[6], h1.z, ag); ag = fmaf((float)g[7], h1.w, ag);
            ao = fmaf((float)o[0], h0.x, ao); ao = fmaf((float)o[1], h0.y, ao);
            ao = fmaf((float)o[2], h0.z, ao); ao = fmaf((float)o[3], h0.w, ao);
            ao = fmaf((float)o[4], h1.x, ao); ao = fmaf((float)o[5], h1.y, ao);
            ao = fmaf((float)o[6], h1.z, ao); ao = fmaf((float)o[7], h1.w, ao);
        }
        float gi = sigm(ai), gf = sigm(af), gg = tanhf(ag), go = sigm(ao);
        c = gf * c + gi * gg;
        float h = go * tanhf(c);
        __syncthreads();              // all dot-product reads of h_lds done
        h_lds[tid] = h;
        hsb[(size_t)t * DH + tid] = (__bf16)h;
        __syncthreads();              // h_lds updated before next step's reads
    }
}

// ---------------- log_softmax over rows of [S, T] ----------------
__global__ __launch_bounds__(DT)
void k_logsoftmax(const float* __restrict__ tag, float* __restrict__ out) {
    __shared__ float red[DT];
    const int row = blockIdx.x, tid = threadIdx.x;
    float x = tag[(size_t)row * DT + tid];
    red[tid] = x;
    __syncthreads();
    for (int s = DT / 2; s > 0; s >>= 1) {
        if (tid < s) red[tid] = fmaxf(red[tid], red[tid + s]);
        __syncthreads();
    }
    float m = red[0];
    __syncthreads();
    red[tid] = __expf(x - m);
    __syncthreads();
    for (int s = DT / 2; s > 0; s >>= 1) {
        if (tid < s) red[tid] += red[tid + s];
        __syncthreads();
    }
    float lse = __logf(red[0]) + m;
    out[(size_t)row * DT + tid] = x - lse;
}

// ---------------- launch ----------------
extern "C" void kernel_launch(void* const* d_in, const int* in_sizes, int n_in,
                              void* d_out, int out_size, void* d_ws, size_t ws_size,
                              hipStream_t stream) {
    (void)in_sizes; (void)n_in; (void)out_size; (void)ws_size;
    const int*   sentence   = (const int*)d_in[0];
    const int*   word_chars = (const int*)d_in[1];
    const float* word_emb   = (const float*)d_in[2];
    const float* char_emb   = (const float*)d_in[3];
    const float* Wih_c      = (const float*)d_in[4];
    const float* Whh_c      = (const float*)d_in[5];
    const float* bih_c      = (const float*)d_in[6];
    const float* bhh_c      = (const float*)d_in[7];
    const float* Wih_t      = (const float*)d_in[8];
    const float* Whh_t      = (const float*)d_in[9];
    const float* bih_t      = (const float*)d_in[10];
    const float* bhh_t      = (const float*)d_in[11];
    const float* W_out      = (const float*)d_in[12];
    const float* b_out      = (const float*)d_in[13];
    float* out = (float*)d_out;

    char* ws = (char*)d_ws;
    size_t off = 0;
    auto take = [&](size_t bytes) -> char* {
        char* p = ws + off;
        off = (off + bytes + 255) & ~(size_t)255;
        return p;
    };
    __bf16* ce     = (__bf16*)take((size_t)DSL * DEC * 2);
    __bf16* wihc_b = (__bf16*)take((size_t)G4HC * DEC * 2);
    __bf16* whhc_b = (__bf16*)take((size_t)G4HC * DHC * 2);
    float*  bias_c = (float*) take((size_t)G4HC * 4);
    float*  xpre_c = (float*) take((size_t)DSL * G4HC * 4);
    __bf16* embeds = (__bf16*)take((size_t)DS * DEHC * 2);
    __bf16* wiht_b = (__bf16*)take((size_t)G4H * DEHC * 2);
    __bf16* whht_b = (__bf16*)take((size_t)G4H * DH * 2);
    float*  bias_t = (float*) take((size_t)G4H * 4);
    float*  xpre_t = (float*) take((size_t)DS * G4H * 4);
    __bf16* hsb    = (__bf16*)take((size_t)DS * DH * 2);
    __bf16* wout_b = (__bf16*)take((size_t)DT * DH * 2);
    float*  tag    = (float*) take((size_t)DS * DT * 4);

    // weight converts + fused biases (bih+bhh folded into the input GEMMs)
    k_convert_bf16<<<256, 256, 0, stream>>>(Wih_c, wihc_b, (long long)G4HC * DEC);
    k_convert_bf16<<<512, 256, 0, stream>>>(Whh_c, whhc_b, (long long)G4HC * DHC);
    k_convert_bf16<<<2048, 256, 0, stream>>>(Wih_t, wiht_b, (long long)G4H * DEHC);
    k_convert_bf16<<<4096, 256, 0, stream>>>(Whh_t, whht_b, (long long)G4H * DH);
    k_convert_bf16<<<256, 256, 0, stream>>>(W_out, wout_b, (long long)DT * DH);
    k_bias_sum<<<(G4HC + 255) / 256, 256, 0, stream>>>(bih_c, bhh_c, bias_c, G4HC);
    k_bias_sum<<<(G4H + 255) / 256, 256, 0, stream>>>(bih_t, bhh_t, bias_t, G4H);

    // embedding gathers
    k_gather_char<<<(DSL * DEC + 255) / 256, 256, 0, stream>>>(word_chars, char_emb, ce);
    k_gather_word<<<(DS * DE + 255) / 256, 256, 0, stream>>>(sentence, word_emb, embeds);

    // xpre_c = ce @ Wih_c.T + (bih_c + bhh_c)      [24576,64]x[64,1024]
    k_gemm_bf16<<<dim3(G4HC / 64, DSL / 64), 128, 0, stream>>>(ce, wihc_b, bias_c, xpre_c,
                                                               DSL, G4HC, DEC);
    // char scan -> embeds[:, 512:768]
    k_char_lstm<<<1, 1024, 0, stream>>>(xpre_c, whhc_b, embeds);

    // xpre_t = embeds @ Wih_t.T + (bih_t + bhh_t)  [2048,768]x[768,4096]
    k_gemm_bf16<<<dim3(G4H / 64, DS / 64), 128, 0, stream>>>(embeds, wiht_b, bias_t, xpre_t,
                                                             DS, G4H, DEHC);
    // word scan -> hs (bf16)
    k_word_lstm<<<1, 1024, 0, stream>>>(xpre_t, whht_b, hsb);

    // tag = hs @ W_out.T + b_out                    [2048,1024]x[1024,128]
    k_gemm_bf16<<<dim3(DT / 64, DS / 64), 128, 0, stream>>>(hsb, wout_b, b_out, tag,
                                                            DS, DT, DH);
    // log_softmax rows
    k_logsoftmax<<<DS, DT, 0, stream>>>(tag, out);
}
